// TTTNeuralCDE_77025943486811
// MI455X (gfx1250) — matrix operand order repaired and checked by reading, not compiled, not run on hardware
//
#include <hip/hip_runtime.h>

typedef _Float16 v16h __attribute__((ext_vector_type(16)));
typedef _Float16 v8h  __attribute__((ext_vector_type(8)));
typedef float    v8f  __attribute__((ext_vector_type(8)));

#define B_   256
#define T_   512
#define C_   16
#define CX_  15
#define H_   128
#define HH_  64
#define MB   16
#define K2_  1056   /* 64*16 outer-product K + 16 bias rows + 16 zero pad */
#define KT2_ 33     /* K2_/32 */
#define EPS_ 1e-5f

#define B2_BYTES (8 * KT2_ * 1024)   /* 270336: dynamic LDS for B' fragments */

__device__ inline v8f wmma16(v16h a, v16h b, v8f c) {
  return __builtin_amdgcn_wmma_f32_16x16x32_f16(false, a, false, b, (short)0, c, false, false);
}

// Packed fragment layout (applies to global packed arrays AND LDS-staged copy):
//   frag f occupies 512 halves: [0..255] = per-lane LOW v8h  at lane*8
//                               [256..511] = per-lane HIGH v8h at lane*8
// 16B lane stride -> conflict-free ds_load_b128.
__device__ inline v16h load_frag(const _Float16* base, int frag, int lane) {
  union { v16h v; v8h h[2]; } u;
  u.h[0] = *(const v8h*)(base + frag * 512 + lane * 8);
  u.h[1] = *(const v8h*)(base + frag * 512 + 256 + lane * 8);
  return u.v;
}

// Load a 16x32 f16 WMMA A-fragment from a row-major LDS matrix (row stride = STRIDE halves).
// lanes 0-15: row=lane, K = 0..7 | 16..23 ; lanes 16-31: row=lane-16, K = 8..15 | 24..31
template<int STRIDE>
__device__ inline v16h load_a(const _Float16* base, int lrow, int lhi, int kk) {
  const _Float16* p = base + lrow * STRIDE + kk * 32 + lhi * 8;
  union { v16h v; v8h h[2]; } u;
  u.h[0] = *(const v8h*)(p);
  u.h[1] = *(const v8h*)(p + 16);
  return u.v;
}

// ---------------------------------------------------------------------------
// Prep: pack W_f1 (128x64), W_a1 (128x64) and the augmented B' (1056x128)
// into f16 WMMA B-fragment order. Fragment f = ntile*KT + kt holds, for
// lane L, column n = ntile*16 + (L&15), rows k = kt*32 + (i>>3)*16 + (L>>4)*8 + (i&7).
// Element i goes to frag*512 + (i>>3)*256 + L*8 + (i&7).
// B'[j*16+c, h] = W_f2[j, h*16+c] ; B'[1024+c, h] = b_f2[h*16+c] ; rest 0.
// ---------------------------------------------------------------------------
__global__ void pack_weights(const float* Wf1, const float* Wa1, const float* Wf2,
                             const float* bf2,
                             _Float16* f1dst, _Float16* a1dst, _Float16* b2dst) {
  int id = blockIdx.x * blockDim.x + threadIdx.x;
  int lane, frag;
  if (id < 1024) {
    const float* src = (id < 512) ? Wf1 : Wa1;
    _Float16* dst = (id < 512) ? f1dst : a1dst;
    id &= 511;
    frag = id >> 5; lane = id & 31;
    int ntile = frag >> 2, kt = frag & 3;
    int n = ntile * 16 + (lane & 15);
    int kbase = kt * 32 + (lane >> 4) * 8;
#pragma unroll
    for (int i = 0; i < 16; i++)
      dst[frag * 512 + (i >> 3) * 256 + lane * 8 + (i & 7)] =
          (_Float16)src[(kbase + (i >> 3) * 16 + (i & 7)) * HH_ + n];
    return;
  }
  id -= 1024;
  if (id >= 8 * KT2_ * 32) return;
  frag = id >> 5; lane = id & 31;
  int ntile = frag / KT2_, kt = frag % KT2_;
  int n = ntile * 16 + (lane & 15);
  int kbase = kt * 32 + (lane >> 4) * 8;
#pragma unroll
  for (int i = 0; i < 16; i++) {
    int kq = kbase + (i >> 3) * 16 + (i & 7);
    float v;
    if (kq < 1024)      v = Wf2[(kq >> 4) * 2048 + n * C_ + (kq & 15)];
    else if (kq < 1040) v = bf2[n * C_ + (kq - 1024)];
    else                v = 0.f;
    b2dst[frag * 512 + (i >> 3) * 256 + lane * 8 + (i & 7)] = (_Float16)v;
  }
}

// ---------------------------------------------------------------------------
// Init: h = x0 @ W_emb + b_emb ; batchnorm over batch ; relu -> z0 (256x128)
// ---------------------------------------------------------------------------
__global__ void init_z0(const float* coeffs, const float* Wemb, const float* bemb,
                        const float* bng, const float* bnb, float* z0) {
  int h = threadIdx.x;
  float sum = 0.f, sumsq = 0.f;
  for (int b = 0; b < B_; b++) {
    float v = bemb[h];
#pragma unroll
    for (int c = 0; c < CX_; c++) v += coeffs[(size_t)b * T_ * C_ + (c + 1)] * Wemb[c * H_ + h];
    sum += v; sumsq += v * v;
  }
  float mu  = sum * (1.f / 256.f);
  float var = sumsq * (1.f / 256.f) - mu * mu;
  float inv = rsqrtf(var + EPS_);
  float g = bng[h], bb = bnb[h];
  for (int b = 0; b < B_; b++) {
    float v = bemb[h];
#pragma unroll
    for (int c = 0; c < CX_; c++) v += coeffs[(size_t)b * T_ * C_ + (c + 1)] * Wemb[c * H_ + h];
    float zn = (v - mu) * inv * g + bb;
    z0[b * H_ + h] = zn > 0.f ? zn : 0.f;
  }
}

// ---------------------------------------------------------------------------
// Persistent scan: 16 blocks x 256 threads (8 wave32s); block owns 16 batch rows.
// B' (264KB) is staged once into dynamic LDS; per-element RK4 state lives in VGPRs.
// ---------------------------------------------------------------------------
__global__ void __launch_bounds__(256) scan_kernel(
    const float* coeffs, const float* bf1, const _Float16* b2fr,
    const _Float16* wf1fr, const _Float16* wa1fr,
    const float* ba1, const float* Wa2, const float* ba2,
    const float* Wo1, const float* bo1, const float* Wo2, const float* bo2,
    const float* Wt1, const float* bt1, const float* Wt2, const float* bt2,
    const float* z0, float* out) {

  extern __shared__ __align__(16) _Float16 ldsB[];          // 8*33*512 halves

  __shared__ __align__(16) float k_s[MB][H_];
  __shared__ __align__(16) _Float16 zin_h[MB][H_];
  __shared__ __align__(16) _Float16 r_h[MB][HH_];
  __shared__ __align__(16) _Float16 dx_h[MB][C_];
  __shared__ float s_s[MB], m_s[MB], l_s[MB];

  const int tid  = threadIdx.x;
  const int wave = tid >> 5;
  const int lane = tid & 31;
  const int lrow = lane & 15;
  const int lhi  = lane >> 4;
  const int bb0  = blockIdx.x * MB;
  const int eb   = tid >> 4;          // element-slice batch row (8 h per thread)
  const int eh0  = (tid & 15) * 8;    // element-slice h base

  // ---- per-element RK4 / attention state in registers ----
  float zr[8], zsumr[8], dr[8], accr[8];
#pragma unroll
  for (int j = 0; j < 8; j++) {
    int h = eh0 + j;
    float v = z0[(bb0 + eb) * H_ + h];
    zr[j] = v; accr[j] = 0.f;
    zin_h[eb][h] = (_Float16)v;
  }
  if (tid < MB) { m_s[tid] = -1e30f; l_s[tid] = 0.f; }

  // ---- stage this wave's 33 B' fragments into LDS (once) ----
  {
    const uint4* gsrc = (const uint4*)((const char*)b2fr + (size_t)wave * KT2_ * 1024);
    uint4*       gdst = (uint4*)((char*)ldsB + (size_t)wave * KT2_ * 1024);
    for (int q = lane; q < KT2_ * 64; q += 32) gdst[q] = gsrc[q];
  }

  // waves 0-3 keep W_f1 / W_a1 fragments resident in VGPRs
  v16h f1b[4], a1b[4];
  float bf1v = 0.f, ba1v = 0.f, wa2v = 0.f;
  if (wave < 4) {
#pragma unroll
    for (int kk = 0; kk < 4; kk++) {
      f1b[kk] = load_frag(wf1fr, wave * 4 + kk, lane);
      a1b[kk] = load_frag(wa1fr, wave * 4 + kk, lane);
    }
    bf1v = bf1[wave * 16 + lrow];
    ba1v = ba1[wave * 16 + lrow];
    wa2v = Wa2[wave * 16 + lrow];
  }
  const float ba2v = ba2[0];
  const _Float16* myB = ldsB + (size_t)wave * KT2_ * 512;   // this wave's n-tile region
  __syncthreads();

  v8h dxh;   // per-lane dx[b=lrow][lhi*8 .. lhi*8+7], refreshed each timestep

  // ---- one ODE-function evaluation: zin_h -> k_s ----
  // g = (A' @ B')  with A'[b, j*16+c] = r[b,j]*dx[b,c], bias K-rows appended.
  auto geval = [&]() {
    // GEMM1: r = relu(z @ W_f1 + b_f1), waves 0-3, each owns one 16-col N tile
    if (wave < 4) {
      v8f acc1 = {};
#pragma unroll
      for (int kk = 0; kk < 4; kk++) {
        v16h a = load_a<H_>(&zin_h[0][0], lrow, lhi, kk);
        acc1 = wmma16(a, f1b[kk], acc1);
      }
#pragma unroll
      for (int i = 0; i < 8; i++) {
        float v = acc1[i] + bf1v;
        v = v > 0.f ? v : 0.f;
        r_h[i + 8 * lhi][wave * 16 + lrow] = (_Float16)v;   // D: VGPR i -> rows i, i+8
      }
    }
    __syncthreads();

    // GEMM2': all 8 waves; wave w owns output columns h = w*16 .. w*16+15.
    v8h rr[8];
    const v8h* rp = (const v8h*)&r_h[lrow][0];
#pragma unroll
    for (int q = 0; q < 8; q++) rr[q] = rp[q];

    v8f acc = {};
#pragma unroll
    for (int kk = 0; kk < 32; kk++) {
      _Float16 r0 = rr[(2 * kk) >> 3][(2 * kk) & 7];
      _Float16 r1 = rr[(2 * kk + 1) >> 3][(2 * kk + 1) & 7];
      union { v16h v; v8h h[2]; } u;
      u.h[0] = dxh * r0;       // A' chunk: j=2kk,   c = lhi*8 .. +7
      u.h[1] = dxh * r1;       // A' chunk: j=2kk+1, c = lhi*8 .. +7
      v16h b = load_frag(myB, kk, lane);
      acc = wmma16(u.v, b, acc);
    }
    { // bias K-tile (kk = 32): A' rows are dx itself (x1), pad rows are zero
      union { v16h v; v8h h[2]; } u;
      v8h zero = {};
      u.h[0] = dxh;
      u.h[1] = zero;
      v16h b = load_frag(myB, 32, lane);
      acc = wmma16(u.v, b, acc);
    }
#pragma unroll
    for (int i = 0; i < 8; i++)
      k_s[i + 8 * lhi][wave * 16 + lrow] = acc[i];   // D tile IS g[b,h]
    __syncthreads();
  };

  // ---- main scan over time: attention(z_t) first, then RK4 -> z_{t+1} ----
  for (int tt = 0; tt < T_; tt++) {
    // attention score for current z: s = tanh(z@W_a1+b_a1)@W_a2 + b_a2 (online softmax)
    if (tid < MB) s_s[tid] = ba2v;
    __syncthreads();
    if (wave < 4) {
      v8f acc = {};
#pragma unroll
      for (int kk = 0; kk < 4; kk++) {
        v16h a = load_a<H_>(&zin_h[0][0], lrow, lhi, kk);
        acc = wmma16(a, a1b[kk], acc);
      }
#pragma unroll
      for (int i = 0; i < 8; i++) {
        float v = tanhf(acc[i] + ba1v) * wa2v;
        v += __shfl_xor(v, 1, 32);
        v += __shfl_xor(v, 2, 32);
        v += __shfl_xor(v, 4, 32);
        v += __shfl_xor(v, 8, 32);
        if (lrow == 0) atomicAdd(&s_s[i + 8 * lhi], v);
      }
    }
    __syncthreads();
    {
      float sv = s_s[eb], mold = m_s[eb], lold = l_s[eb];
      __syncthreads();
      float mnew = fmaxf(mold, sv);
      float f = __expf(mold - mnew), e = __expf(sv - mnew);
#pragma unroll
      for (int j = 0; j < 8; j++) accr[j] = accr[j] * f + e * zr[j];
      if ((tid & 15) == 0) { m_s[eb] = mnew; l_s[eb] = lold * f + e; }
      __syncthreads();
    }
    if (tt == T_ - 1) break;

    // dx for this step (+ prefetch the next step's coeffs line)
    {
      int b = tid >> 4, c = tid & 15;
      const float* cp = coeffs + (size_t)(bb0 + b) * T_ * C_ + (size_t)tt * C_ + c;
      dx_h[b][c] = (_Float16)(cp[C_] - cp[0]);
      __builtin_prefetch(cp + 2 * C_, 0, 0);
    }
    __syncthreads();
    dxh = *(const v8h*)&dx_h[lrow][lhi * 8];

    // ---- RK4 ----
    geval();   // k1
#pragma unroll
    for (int j = 0; j < 8; j++) {
      float k1v = k_s[eb][eh0 + j];
      zsumr[j] = k1v;
      dr[j] = k1v;
      zin_h[eb][eh0 + j] = (_Float16)(zr[j] + k1v * (1.f / 3.f));
    }
    __syncthreads();
    geval();   // k2
#pragma unroll
    for (int j = 0; j < 8; j++) {
      float k2v = k_s[eb][eh0 + j];
      float k1v = dr[j];
      zin_h[eb][eh0 + j] = (_Float16)(zr[j] + k2v - k1v * (1.f / 3.f));
      dr[j] = k1v - k2v;
      zsumr[j] += 3.f * k2v;
    }
    __syncthreads();
    geval();   // k3
#pragma unroll
    for (int j = 0; j < 8; j++) {
      float k3v = k_s[eb][eh0 + j];
      zin_h[eb][eh0 + j] = (_Float16)(zr[j] + dr[j] + k3v);
      zsumr[j] += 3.f * k3v;
    }
    __syncthreads();
    geval();   // k4
#pragma unroll
    for (int j = 0; j < 8; j++) {
      float zn = zr[j] + (zsumr[j] + k_s[eb][eh0 + j]) * 0.125f;
      zr[j] = zn;
      zin_h[eb][eh0 + j] = (_Float16)zn;
    }
    __syncthreads();
  }

  // ---- epilogue: z_hat = acc/l ; output heads ----
  {
    float linv = 1.f / l_s[eb];
#pragma unroll
    for (int j = 0; j < 8; j++) {
      int h = eh0 + j;
      float zh = accr[j] * linv;
      k_s[eb][h] = zh;   // k_s now holds z_hat (f32) for the heads
      out[2304 + (size_t)(bb0 + eb) * H_ + h] = zh;
    }
  }
  __syncthreads();
  if (tid < MB) {
    int b = tid, bg = bb0 + b;
    float y = bo2[0];
    for (int k = 0; k < HH_; k++) {
      float hv = bo1[k];
      for (int h = 0; h < H_; h++) hv += k_s[b][h] * Wo1[h * HH_ + k];
      hv = fmaxf(hv, 0.f);
      y += hv * Wo2[k];
    }
    out[bg] = y;
    float pa[4];
#pragma unroll
    for (int j2 = 0; j2 < 4; j2++) pa[j2] = bt2[j2];
    for (int k = 0; k < HH_; k++) {
      float hv = bt1[k];
      for (int h = 0; h < H_; h++) hv += k_s[b][h] * Wt1[h * HH_ + k];
      hv = fmaxf(hv, 0.f);
#pragma unroll
      for (int j2 = 0; j2 < 4; j2++) pa[j2] += hv * Wt2[k * 4 + j2];
    }
    float mx = fmaxf(fmaxf(pa[0], pa[1]), fmaxf(pa[2], pa[3]));
    float ex[4], se = 0.f;
#pragma unroll
    for (int j2 = 0; j2 < 4; j2++) { ex[j2] = __expf(pa[j2] - mx); se += ex[j2]; }
    float sinv = 1.f / se;
#pragma unroll
    for (int j2 = 0; j2 < 4; j2++) {
      out[256 + bg * 4 + j2]  = ex[j2] * sinv;   // pred_a_softmax
      out[1280 + bg * 4 + j2] = pa[j2];          // pred_a
    }
  }
}

// ---------------------------------------------------------------------------
extern "C" void kernel_launch(void* const* d_in, const int* in_sizes, int n_in,
                              void* d_out, int out_size, void* d_ws, size_t ws_size,
                              hipStream_t stream) {
  (void)in_sizes; (void)n_in; (void)out_size; (void)ws_size;
  const float* coeffs = (const float*)d_in[0];
  const float* Wemb = (const float*)d_in[1];
  const float* bemb = (const float*)d_in[2];
  const float* bng  = (const float*)d_in[3];
  const float* bnb  = (const float*)d_in[4];
  const float* Wf1  = (const float*)d_in[5];
  const float* bf1  = (const float*)d_in[6];
  const float* Wf2  = (const float*)d_in[7];
  const float* bf2  = (const float*)d_in[8];
  const float* Wa1  = (const float*)d_in[9];
  const float* ba1  = (const float*)d_in[10];
  const float* Wa2  = (const float*)d_in[11];
  const float* ba2  = (const float*)d_in[12];
  const float* Wo1  = (const float*)d_in[13];
  const float* bo1  = (const float*)d_in[14];
  const float* Wo2  = (const float*)d_in[15];
  const float* bo2  = (const float*)d_in[16];
  const float* Wt1  = (const float*)d_in[17];
  const float* bt1  = (const float*)d_in[18];
  const float* Wt2  = (const float*)d_in[19];
  const float* bt2  = (const float*)d_in[20];

  char* ws = (char*)d_ws;
  float*    z0   = (float*)(ws);              // 131072 B
  _Float16* f1fr = (_Float16*)(ws + 131072);  // 16384 B
  _Float16* a1fr = (_Float16*)(ws + 147456);  // 16384 B
  _Float16* b2fr = (_Float16*)(ws + 163840);  // 8*33*1024 = 270336 B

  // allow 264KB of dynamic LDS (gfx1250: up to 320KB per workgroup)
  (void)hipFuncSetAttribute((const void*)scan_kernel,
                            hipFuncAttributeMaxDynamicSharedMemorySize, B2_BYTES);

  pack_weights<<<dim3(37), dim3(256), 0, stream>>>(Wf1, Wa1, Wf2, bf2, f1fr, a1fr, b2fr);
  init_z0<<<dim3(1), dim3(128), 0, stream>>>(coeffs, Wemb, bemb, bng, bnb, z0);
  scan_kernel<<<dim3(16), dim3(256), B2_BYTES, stream>>>(
      coeffs, bf1, b2fr, f1fr, a1fr, ba1, Wa2, ba2,
      Wo1, bo1, Wo2, bo2, Wt1, bt1, Wt2, bt2, z0, (float*)d_out);
}
